// TransformerBlock_69509750718529
// MI455X (gfx1250) — compile-verified, matrix operand order checked by reading, not run on hardware
//
#include <hip/hip_runtime.h>
#include <cstdint>
#include <cstddef>

typedef __attribute__((ext_vector_type(16))) __bf16 v16bf;
typedef __attribute__((ext_vector_type(8)))  __bf16 v8bf;
typedef __attribute__((ext_vector_type(4)))  __bf16 v4bf;
typedef __attribute__((ext_vector_type(8)))  float  v8f;

static constexpr int    EMB   = 512;
static constexpr int    HIDD  = 2048;
static constexpr size_t MTOT  = 32768;   // G*L
static constexpr int    TLDS  = 40;      // padded LDS stride (bf16) for 32-wide K tiles

// ---------------------------------------------------------------------------
// CDNA5 async global->LDS copy (GLOBAL_LOAD_ASYNC_TO_LDS_B128, ASYNCcnt) with
// graceful fallback to a sync VGPR round-trip when the builtin is absent.
// Builtin prototype (from hipcc diagnostic): first param is a pointer to
// gcc-vector int4 in AS1 (printed "__device__"), so cast to exactly that.
// ---------------------------------------------------------------------------
#if defined(__gfx1250__) && __has_builtin(__builtin_amdgcn_global_load_async_to_lds_b128)
#define ASYNC_LDS 1
#else
#define ASYNC_LDS 0
#endif

typedef int v4i_raw __attribute__((vector_size(16)));
typedef __attribute__((address_space(1))) v4i_raw* g128p;   // global int4*
typedef __attribute__((address_space(3))) v4i_raw* l128p;   // LDS int4*

static __device__ __forceinline__ void cp16_to_lds(const __bf16* g, __bf16* l) {
#if ASYNC_LDS
  __builtin_amdgcn_global_load_async_to_lds_b128((g128p)g, (l128p)l, 0, 0);
#else
  *(v8bf*)l = *(const v8bf*)g;
#endif
}

static __device__ __forceinline__ void async_wait_all() {
#if ASYNC_LDS
#if __has_builtin(__builtin_amdgcn_s_wait_asynccnt)
  __builtin_amdgcn_s_wait_asynccnt(0);
#else
  asm volatile("s_wait_asynccnt 0" ::: "memory");
#endif
#endif
}

// ---------------------------------------------------------------------------
// Fragment load from LDS following CDNA5 16-bit A/B layout:
//   lanes 0-15  : row = lane,    K = {0..7, 16..23}
//   lanes 16-31 : row = lane-16, K = {8..15, 24..31}
// => two contiguous 16B reads per lane.
// ---------------------------------------------------------------------------
static __device__ __forceinline__ v16bf frag_ld(const __bf16* s, int row0, int kofs, int stride) {
  const int lane = threadIdx.x & 31;
  const __bf16* p = s + (size_t)(row0 + (lane & 15)) * stride + kofs + ((lane >> 4) & 1) * 8;
  v8bf lo = *(const v8bf*)p;
  v8bf hi = *(const v8bf*)(p + 16);
  return __builtin_shufflevector(lo, hi, 0,1,2,3,4,5,6,7,8,9,10,11,12,13,14,15);
}

static __device__ __forceinline__ v8f wmma_bf16(v16bf a, v16bf b, v8f c) {
  return __builtin_amdgcn_wmma_f32_16x16x32_bf16(false, a, false, b, (short)0, c, false, false);
}

// ---------------------------------------------------------------------------
// GEMM: C[M x N] = A[M x K] * W[N x K]^T + bias, tiled 128x128 per block.
// 8 waves (4 along M, 2 along N); each wave: 32x64 output = 2x4 WMMA frags.
// Double-buffered LDS: tile k+1 is staged (async) while tile k is consumed.
// ACT: SiLU. RES: add f32 residual. OUTF32: store f32 (else bf16).
// ---------------------------------------------------------------------------
template<int ACT, int RES, int OUTF32>
__global__ void __launch_bounds__(256)
gemm_bf16_k(const __bf16* __restrict__ A, int lda,
            const __bf16* __restrict__ W, int ldw,
            const float*  __restrict__ bias,
            float*  __restrict__ Cf, __bf16* __restrict__ Cb, int ldc,
            const float* __restrict__ resid, int K)
{
  __shared__ __bf16 sA[2][128 * TLDS];
  __shared__ __bf16 sB[2][128 * TLDS];

  const int t    = threadIdx.x;
  const int bm   = blockIdx.y * 128;
  const int bn   = blockIdx.x * 128;
  const int wave = t >> 5;
  const int wm   = wave & 3;   // 0..3 along M (32 rows each)
  const int wn   = wave >> 2;  // 0..1 along N (64 cols each)

  // Stage one 128x32 tile pair; 512 16B-chunks per matrix, 2 per thread.
  auto stage = [&](int k0, int b) {
    #pragma unroll
    for (int i = 0; i < 2; ++i) {
      const int chunk = t + i * 256;
      const int row   = chunk >> 2;
      const int cc    = (chunk & 3) * 8;
      cp16_to_lds(A + (size_t)(bm + row) * lda + k0 + cc, &sA[b][row * TLDS + cc]);
      cp16_to_lds(W + (size_t)(bn + row) * ldw + k0 + cc, &sB[b][row * TLDS + cc]);
    }
  };

  const v8f vzero = {0.f,0.f,0.f,0.f,0.f,0.f,0.f,0.f};
  v8f acc[2][4];
  #pragma unroll
  for (int mt = 0; mt < 2; ++mt)
    #pragma unroll
    for (int nt = 0; nt < 4; ++nt)
      acc[mt][nt] = vzero;

  stage(0, 0);
  async_wait_all();
  __syncthreads();

  int buf = 0;
  for (int k0 = 0; k0 < K; k0 += 32) {
    const bool more = (k0 + 32 < K);
    if (more) stage(k0 + 32, buf ^ 1);           // overlap next-tile staging
    if (k0 + 64 < K) {                            // prefetch 2 tiles ahead -> GL2
      const int prow = t >> 1;
      const int pcc  = (t & 1) * 16;
      __builtin_prefetch(A + (size_t)(bm + prow) * lda + k0 + 64 + pcc, 0, 3);
      __builtin_prefetch(W + (size_t)(bn + prow) * ldw + k0 + 64 + pcc, 0, 3);
    }

    const __bf16* cA = sA[buf];
    const __bf16* cB = sB[buf];
    v16bf af[2], bf[4];
    af[0] = frag_ld(cA, wm * 32,      0, TLDS);
    af[1] = frag_ld(cA, wm * 32 + 16, 0, TLDS);
    #pragma unroll
    for (int nt = 0; nt < 4; ++nt) bf[nt] = frag_ld(cB, wn * 64 + nt * 16, 0, TLDS);

    #pragma unroll
    for (int mt = 0; mt < 2; ++mt)
      #pragma unroll
      for (int nt = 0; nt < 4; ++nt)
        acc[mt][nt] = wmma_bf16(af[mt], bf[nt], acc[mt][nt]);

    if (more) async_wait_all();
    __syncthreads();
    buf ^= 1;
  }

  // Epilogue. C/D frag layout: vgpr i -> row = i + 8*(lane>=16), col = lane&15.
  const int lane  = t & 31;
  const int half8 = ((lane >> 4) & 1) * 8;
  const int nl    = lane & 15;
  #pragma unroll
  for (int mt = 0; mt < 2; ++mt) {
    #pragma unroll
    for (int nt = 0; nt < 4; ++nt) {
      const int   col = bn + wn * 64 + nt * 16 + nl;
      const float bv  = bias[col];
      #pragma unroll
      for (int i = 0; i < 8; ++i) {
        const int row = bm + wm * 32 + mt * 16 + half8 + i;
        float v = acc[mt][nt][i] + bv;
        if (ACT) v = v / (1.f + __expf(-v));             // SiLU
        const size_t idx = (size_t)row * ldc + col;
        if (RES) v += resid[idx];
        if (OUTF32) Cf[idx] = v;
        else        Cb[idx] = (__bf16)v;
      }
    }
  }
}

// ---------------------------------------------------------------------------
// LayerNorm over 512 f32 -> bf16. One block (128 threads, 4 elems each) per row.
// ---------------------------------------------------------------------------
__global__ void __launch_bounds__(128)
layernorm_k(const float* __restrict__ X, const float* __restrict__ g,
            const float* __restrict__ b, __bf16* __restrict__ Y)
{
  __shared__ float r1[128], r2[128];
  const int row = blockIdx.x;
  const int t   = threadIdx.x;
  const float4 v = ((const float4*)(X + (size_t)row * EMB))[t];
  r1[t] = v.x + v.y + v.z + v.w;
  r2[t] = v.x*v.x + v.y*v.y + v.z*v.z + v.w*v.w;
  __syncthreads();
  for (int o = 64; o > 0; o >>= 1) {
    if (t < o) { r1[t] += r1[t + o]; r2[t] += r2[t + o]; }
    __syncthreads();
  }
  const float mu = r1[0] * (1.f / EMB);
  const float var = r2[0] * (1.f / EMB) - mu * mu;
  const float rs = rsqrtf(var + 1e-5f);
  const int c = t * 4;
  v4bf o4;
  o4[0] = (__bf16)((v.x - mu) * rs * g[c + 0] + b[c + 0]);
  o4[1] = (__bf16)((v.y - mu) * rs * g[c + 1] + b[c + 1]);
  o4[2] = (__bf16)((v.z - mu) * rs * g[c + 2] + b[c + 2]);
  o4[3] = (__bf16)((v.w - mu) * rs * g[c + 3] + b[c + 3]);
  *(v4bf*)(Y + (size_t)row * EMB + c) = o4;
}

// ---------------------------------------------------------------------------
// Flash-style attention. Block = 256 threads handles one (group, head, 64-row
// query block). d = 64, L = 512. S = Q K^T and O += P V via WMMA; streaming
// softmax state (m, l) in LDS. P reuses the K tile to stay under 64KB LDS.
// Q/K tiles staged via async-to-LDS; V is transposed through VGPRs.
// ---------------------------------------------------------------------------
__global__ void __launch_bounds__(256)
attn_k(const __bf16* __restrict__ QKV, __bf16* __restrict__ O)
{
  __shared__ __bf16 sQ [64 * 72];
  __shared__ __bf16 sK [64 * 72];   // reused as P (bf16) after softmax
  __shared__ __bf16 sVT[64 * 72];   // V transposed: [d][key]
  __shared__ float  sS [64 * 68];
  __shared__ float  sO [64 * 68];
  __shared__ float  sM[64], sL[64], sAl[64];

  const int t  = threadIdx.x;
  const int qb = blockIdx.x;   // 0..7  query block
  const int h  = blockIdx.y;   // 0..7  head
  const int g  = blockIdx.z;   // 0..63 group
  const size_t base = (size_t)g * 512 * 1536;
  const __bf16* Qg = QKV + base + h * 64;
  const __bf16* Kg = QKV + base + 512 + h * 64;
  const __bf16* Vg = QKV + base + 1024 + h * 64;

  // Load Q block (64x64): 512 16B chunks, 2 per thread (async).
  #pragma unroll
  for (int i = 0; i < 2; ++i) {
    const int chunk = t + i * 256;
    const int row = chunk >> 3;
    const int cc  = (chunk & 7) * 8;
    cp16_to_lds(Qg + (size_t)(qb * 64 + row) * 1536 + cc, sQ + row * 72 + cc);
  }
  #pragma unroll
  for (int e = 0; e < 16; ++e) {
    const int idx = t + e * 256;
    sO[(idx >> 6) * 68 + (idx & 63)] = 0.f;
  }
  if (t < 64) { sM[t] = -1e30f; sL[t] = 0.f; }

  const int wave  = t >> 5;
  const int wm    = wave & 1;    // 2 along rows (32 each)
  const int wn    = wave >> 1;   // 4 along cols (16 each)
  const int lane  = t & 31;
  const int half8 = ((lane >> 4) & 1) * 8;
  const int nl    = lane & 15;
  const v8f vzero = {0.f,0.f,0.f,0.f,0.f,0.f,0.f,0.f};

  for (int j = 0; j < 8; ++j) {
    // Load K block (async) and V^T block (VGPR transpose scatter).
    #pragma unroll
    for (int i = 0; i < 2; ++i) {
      const int chunk = t + i * 256;
      const int row = chunk >> 3;          // key index
      const int cc  = (chunk & 7) * 8;     // d offset
      cp16_to_lds(Kg + (size_t)(j * 64 + row) * 1536 + cc, sK + row * 72 + cc);
      v8bf vv = *(const v8bf*)(Vg + (size_t)(j * 64 + row) * 1536 + cc);
      #pragma unroll
      for (int e = 0; e < 8; ++e) sVT[(cc + e) * 72 + row] = vv[e];
    }
    async_wait_all();
    __syncthreads();

    // S = Q K^T * (1/sqrt(64))
    {
      v8f a0 = vzero, a1 = vzero;
      #pragma unroll
      for (int ks = 0; ks < 64; ks += 32) {
        v16bf bq = frag_ld(sK, wn * 16, ks, 72);
        a0 = wmma_bf16(frag_ld(sQ, wm * 32,      ks, 72), bq, a0);
        a1 = wmma_bf16(frag_ld(sQ, wm * 32 + 16, ks, 72), bq, a1);
      }
      #pragma unroll
      for (int i = 0; i < 8; ++i) {
        sS[(wm * 32 +      half8 + i) * 68 + wn * 16 + nl] = a0[i] * 0.125f;
        sS[(wm * 32 + 16 + half8 + i) * 68 + wn * 16 + nl] = a1[i] * 0.125f;
      }
    }
    __syncthreads();

    // Streaming softmax per row; write P (bf16) over sK.
    if (t < 64) {
      const float mo = sM[t];
      float mx = mo;
      for (int c = 0; c < 64; ++c) mx = fmaxf(mx, sS[t * 68 + c]);
      const float al = __expf(mo - mx);
      float l = sL[t] * al;
      for (int c = 0; c < 64; ++c) {
        const float p = __expf(sS[t * 68 + c] - mx);
        l += p;
        sK[t * 72 + c] = (__bf16)p;
      }
      sM[t] = mx; sL[t] = l; sAl[t] = al;
    }
    __syncthreads();

    // Rescale O accumulator.
    #pragma unroll
    for (int e = 0; e < 16; ++e) {
      const int idx = t + e * 256;
      sO[(idx >> 6) * 68 + (idx & 63)] *= sAl[idx >> 6];
    }
    __syncthreads();

    // O += P V  (K-dim = keys)
    {
      v8f a0 = vzero, a1 = vzero;
      #pragma unroll
      for (int ks = 0; ks < 64; ks += 32) {
        v16bf bv = frag_ld(sVT, wn * 16, ks, 72);
        a0 = wmma_bf16(frag_ld(sK, wm * 32,      ks, 72), bv, a0);
        a1 = wmma_bf16(frag_ld(sK, wm * 32 + 16, ks, 72), bv, a1);
      }
      #pragma unroll
      for (int i = 0; i < 8; ++i) {
        sO[(wm * 32 +      half8 + i) * 68 + wn * 16 + nl] += a0[i];
        sO[(wm * 32 + 16 + half8 + i) * 68 + wn * 16 + nl] += a1[i];
      }
    }
    __syncthreads();
  }

  // Normalize and write out.
  #pragma unroll
  for (int e = 0; e < 16; ++e) {
    const int idx = t + e * 256;
    const int r = idx >> 6, c = idx & 63;
    const float val = sO[r * 68 + c] / sL[r];
    O[(size_t)(g * 512 + qb * 64 + r) * 512 + h * 64 + c] = (__bf16)val;
  }
}

// ---------------------------------------------------------------------------
// Small helpers: f32->bf16 convert, feature slice init, final 2048->1 dot.
// ---------------------------------------------------------------------------
__global__ void cvt_k(const float* __restrict__ src, __bf16* __restrict__ dst, int n4)
{
  const int i = blockIdx.x * blockDim.x + threadIdx.x;
  if (i < n4) {
    const float4 v = ((const float4*)src)[i];
    v4bf o;
    o[0] = (__bf16)v.x; o[1] = (__bf16)v.y; o[2] = (__bf16)v.z; o[3] = (__bf16)v.w;
    ((v4bf*)dst)[i] = o;
  }
}

__global__ void initx_k(const float* __restrict__ F, float* __restrict__ X)
{
  const size_t idx = (size_t)blockIdx.x * blockDim.x + threadIdx.x; // over MTOT*128 float4
  const size_t r = idx >> 7, c4 = idx & 127;
  ((float4*)X)[r * 128 + c4] = ((const float4*)F)[r * 224 + c4];    // 896 f32 = 224 float4
}

__global__ void __launch_bounds__(256)
fdot_k(const __bf16* __restrict__ Hb, const __bf16* __restrict__ w2,
       const float* __restrict__ b2, float* __restrict__ out)
{
  const int row  = blockIdx.x * 8 + (threadIdx.x >> 5);
  const int lane = threadIdx.x & 31;
  const __bf16* hr = Hb + (size_t)row * HIDD;
  float s = 0.f;
  #pragma unroll
  for (int i = 0; i < 8; ++i) {
    const int o = (i * 32 + lane) * 8;
    v8bf a = *(const v8bf*)(hr + o);
    v8bf w = *(const v8bf*)(w2 + o);
    #pragma unroll
    for (int e = 0; e < 8; ++e) s += (float)a[e] * (float)w[e];
  }
  #pragma unroll
  for (int m = 16; m > 0; m >>= 1) s += __shfl_xor(s, m, 32);
  if (lane == 0) out[row] = s + b2[0];
}

// ---------------------------------------------------------------------------
// Host orchestration.
// Input order (dict flatten): 0 features, 1 ensemble_index, then per group:
//  kqv{g,b,w1,b1,w2,b2} attn{w_in,b_in,w_out,b_out} ff{g,b,w1,b1,w2,b2}
//  -> kqv_k at 2+16k, attn_k at 8+16k, ff_k at 12+16k (k=0,1), final at 44.
// ---------------------------------------------------------------------------
extern "C" void kernel_launch(void* const* d_in, const int* in_sizes, int n_in,
                              void* d_out, int out_size, void* d_ws, size_t ws_size,
                              hipStream_t stream)
{
  (void)in_sizes; (void)out_size; (void)ws_size;
  const float* P[64];
  for (int i = 0; i < n_in && i < 64; ++i) P[i] = (const float*)d_in[i];

  char* wp = (char*)d_ws;
  auto walloc = [&](size_t bytes) -> void* {
    void* p = (void*)wp; wp += (bytes + 255) & ~(size_t)255; return p;
  };
  float*  X    = (float*) walloc(MTOT * EMB  * 4);
  __bf16* LN   = (__bf16*)walloc(MTOT * EMB  * 2);
  __bf16* Hbuf = (__bf16*)walloc(MTOT * HIDD * 2);
  __bf16* QKV  = (__bf16*)walloc(MTOT * 1536 * 2);
  __bf16* QKV2 = (__bf16*)walloc(MTOT * 1536 * 2);
  __bf16* Obuf = (__bf16*)walloc(MTOT * EMB  * 2);

  __bf16 *kw1[3], *kw2[3], *awin[3], *awout[3], *fw1[2], *fw2[2], *gw1, *gw2;
  for (int k = 0; k < 3; ++k) {
    kw1[k]   = (__bf16*)walloc((size_t)HIDD * EMB * 2);
    kw2[k]   = (__bf16*)walloc((size_t)1536 * HIDD * 2);
    awin[k]  = (__bf16*)walloc((size_t)1536 * EMB * 2);
    awout[k] = (__bf16*)walloc((size_t)EMB * EMB * 2);
  }
  for (int k = 0; k < 2; ++k) {
    fw1[k] = (__bf16*)walloc((size_t)HIDD * EMB * 2);
    fw2[k] = (__bf16*)walloc((size_t)EMB * HIDD * 2);
  }
  gw1 = (__bf16*)walloc((size_t)HIDD * EMB * 2);
  gw2 = (__bf16*)walloc((size_t)HIDD * 2);

  auto cvt = [&](const float* s, __bf16* d, size_t n) {
    const int n4 = (int)(n / 4);
    cvt_k<<<(n4 + 255) / 256, 256, 0, stream>>>(s, d, n4);
  };
  for (int k = 0; k < 3; ++k) {
    const int pk = 2 + k * 16, pa = 8 + k * 16;
    cvt(P[pk + 2], kw1[k],  (size_t)HIDD * EMB);
    cvt(P[pk + 4], kw2[k],  (size_t)1536 * HIDD);
    cvt(P[pa],     awin[k], (size_t)1536 * EMB);
    cvt(P[pa + 2], awout[k],(size_t)EMB * EMB);
  }
  for (int k = 0; k < 2; ++k) {
    const int pf = 12 + k * 16;
    cvt(P[pf + 2], fw1[k], (size_t)HIDD * EMB);
    cvt(P[pf + 4], fw2[k], (size_t)EMB * HIDD);
  }
  cvt(P[46], gw1, (size_t)HIDD * EMB);
  cvt(P[48], gw2, (size_t)HIDD);

  initx_k<<<(unsigned)((MTOT * 128) / 256), 256, 0, stream>>>(P[0], X);

  const dim3 blk(256);
  auto ggrid = [&](int N) { return dim3((unsigned)(N / 128), (unsigned)(MTOT / 128)); };

  for (int k = 0; k < 3; ++k) {
    const int pk = 2 + k * 16, pa = 8 + k * 16;
    // kqv MLP: LN -> silu(LN@w1^T+b1) -> (@w2^T+b2) -> QKV (c1|c2|c3)
    layernorm_k<<<(unsigned)MTOT, 128, 0, stream>>>(X, P[pk], P[pk + 1], LN);
    gemm_bf16_k<1,0,0><<<ggrid(HIDD), blk, 0, stream>>>(LN, EMB, kw1[k], EMB, P[pk + 3],
                                                        nullptr, Hbuf, HIDD, nullptr, EMB);
    gemm_bf16_k<0,0,0><<<ggrid(1536), blk, 0, stream>>>(Hbuf, HIDD, kw2[k], HIDD, P[pk + 5],
                                                        nullptr, QKV, 1536, nullptr, HIDD);
    // q/k/v projections from c1/c2/c3
    for (int s = 0; s < 3; ++s)
      gemm_bf16_k<0,0,0><<<ggrid(EMB), blk, 0, stream>>>(QKV + s * EMB, 1536,
          awin[k] + (size_t)s * EMB * EMB, EMB, P[pa + 1] + s * EMB,
          nullptr, QKV2 + s * EMB, 1536, nullptr, EMB);
    // attention
    attn_k<<<dim3(8, 8, 64), blk, 0, stream>>>(QKV2, Obuf);
    // out projection -> X (f32, replaces residual stream)
    gemm_bf16_k<0,0,1><<<ggrid(EMB), blk, 0, stream>>>(Obuf, EMB, awout[k], EMB, P[pa + 3],
                                                       X, nullptr, EMB, nullptr, EMB);
    if (k < 2) {
      const int pf = 12 + k * 16;
      layernorm_k<<<(unsigned)MTOT, 128, 0, stream>>>(X, P[pf], P[pf + 1], LN);
      gemm_bf16_k<1,0,0><<<ggrid(HIDD), blk, 0, stream>>>(LN, EMB, fw1[k], EMB, P[pf + 3],
                                                          nullptr, Hbuf, HIDD, nullptr, EMB);
      gemm_bf16_k<0,1,1><<<ggrid(EMB), blk, 0, stream>>>(Hbuf, HIDD, fw2[k], HIDD, P[pf + 5],
                                                         X, nullptr, EMB, X, HIDD);
    }
  }
  // final head: LN -> silu(@w1^T+b1) -> dot with w2 (+b2)
  layernorm_k<<<(unsigned)MTOT, 128, 0, stream>>>(X, P[44], P[45], LN);
  gemm_bf16_k<1,0,0><<<ggrid(HIDD), blk, 0, stream>>>(LN, EMB, gw1, EMB, P[47],
                                                      nullptr, Hbuf, HIDD, nullptr, EMB);
  fdot_k<<<(unsigned)(MTOT / 8), 256, 0, stream>>>(Hbuf, gw2, P[49], (float*)d_out);
}
